// NMODEL_27633819582859
// MI455X (gfx1250) — matrix operand order repaired and checked by reading, not compiled, not run on hardware
//
#include <hip/hip_runtime.h>

// MI455X / gfx1250, wave32. Sequential neural-ODE RNN.
// 64 single-wave workgroups, 16 batch rows each; h state in VGPRs;
// Wf f16 B-fragments in LDS; MLP weights as pre-swizzled f16 B-fragment
// streams in d_ws (L2-resident); f16 WMMA with f32 accumulation.
// RK4 stages consume k tile-by-tile (no k/ks arrays) and stage the next
// stage's A operand into LDS as they go. A per-iteration opaque scalar
// offset defeats LICM (no hoist->spill) without breaking address-space
// inference (loads stay global_load/ds_load, not flat).

#define MODES  64
#define NB     1024
#define NT     128
#define NSTEPS 8

typedef __attribute__((ext_vector_type(16))) _Float16 h16x16; // WMMA A/B fragment (8 VGPRs)
typedef __attribute__((ext_vector_type(8)))  _Float16 h16x8;
typedef __attribute__((ext_vector_type(8)))  float    v8f;    // WMMA C/D fragment
typedef __attribute__((ext_vector_type(4)))  float    v4f;

#define SB() __builtin_amdgcn_sched_barrier(0)

// zero-cost LICM breaker: opaque scalar zero, redefined at each call site
static __device__ __forceinline__ int opaque0() {
  int z = 0;
  asm volatile("" : "+s"(z));
  return z;
}

static __device__ __forceinline__ v8f splat8(float x) {
  v8f v;
#pragma unroll
  for (int i = 0; i < 8; i++) v[i] = x;
  return v;
}
static __device__ __forceinline__ float sigm(float x) { return 1.f / (1.f + __expf(-x)); }
static __device__ __forceinline__ float tanh_f(float x) {
  float e = __expf(-2.f * fabsf(x));
  float r = (1.f - e) / (1.f + e);
  return x < 0.f ? -r : r;
}

static __device__ __forceinline__ v8f wmma16(h16x16 a, h16x16 b, v8f c) {
  // (neg_a, A, neg_b, B, c_mod, C, reuse_a, reuse_b)
  return __builtin_amdgcn_wmma_f32_16x16x32_f16(false, a, false, b, (short)0, c, false, false);
}

// B fragment stream element order matches ISA B layout:
// lane L: N = nt*16 + (L&15), element e: K = kc*32 + 8*(L>>4) + (e&7) + 16*(e>>3)
static __device__ __forceinline__ h16x16 loadB(const _Float16* base, int ti, int lane) {
  return *(const h16x16*)(base + ti * 512 + lane * 16);
}

// A fragment read from ldsA[m][K] row-major (row stride 256 elements):
// lane L: M = L&15, e<8: K = c*32 + 8*(L>>4) + e ; e>=8: K += 16
static __device__ __forceinline__ h16x16 loadA(const _Float16* rowbase, int c, int HL) {
  h16x8 lo = *(const h16x8*)(rowbase + c * 32 + 8 * HL);
  h16x8 hi = *(const h16x8*)(rowbase + c * 32 + 16 + 8 * HL);
  h16x16 v;
#pragma unroll
  for (int i = 0; i < 8; i++) { v[i] = lo[i]; v[i + 8] = hi[i]; }
  return v;
}

// Scatter a D-layout f32 tile (M = 8*HL + r, N = nt*16 + ls) into ldsA[m][K]
static __device__ __forceinline__ void storeD(_Float16* lds, int nt, int HL, int ls, v8f d) {
  _Float16* p = lds + (8 * HL) * 256 + nt * 16 + ls;
#pragma unroll
  for (int r = 0; r < 8; r++) p[r * 256] = (_Float16)d[r];
}

// One RK4 stage: ldsA holds the stage input (16x128). Computes
// k = c*sigmoid(in @ Wf + bf) tile-by-tile; folds into hn with weight wacc;
// writes next stage input (h + alpha_next*k, or new h if LAST) back to ldsA.
template<bool FIRST, bool LAST>
static __device__ __forceinline__ void rkstage(
    _Float16* ldsA, const _Float16* ldsWf_in,
    v8f* h, v8f* hn, float wacc, float alpha_next,
    const float* cr, const float* bf_l, int HL, int ls, int L)
{
  const _Float16* ldsWf = ldsWf_in + opaque0();  // invariant across 32 stages: keep un-hoisted
  h16x16 ach[4];
#pragma unroll
  for (int c = 0; c < 4; c++) ach[c] = loadA(ldsA + ls * 256, c, HL);
  SB();
#pragma unroll
  for (int nt = 0; nt < 8; nt++) {
    v8f acc = splat8(bf_l[nt]);
#pragma unroll
    for (int c = 0; c < 4; c++)
      acc = wmma16(ach[c], loadB(ldsWf, nt * 4 + c, L), acc);
    v8f kv;
#pragma unroll
    for (int r = 0; r < 8; r++) kv[r] = cr[r] * sigm(acc[r]);
    if (FIRST) hn[nt] = h[nt] + wacc * kv;
    else       hn[nt] += wacc * kv;
    v8f nxt;
    if (LAST) { h[nt] = hn[nt]; nxt = hn[nt]; }
    else      { nxt = h[nt] + alpha_next * kv; }
    storeD(ldsA, nt, HL, ls, nxt);
    if (nt & 1) SB();   // bound scheduler lookahead to 2 tiles
  }
  __syncthreads();      // single-wave workgroup: hardware NOP, ordering fence
}

// ---- prep: swizzle weights into f16 B-fragment streams in d_ws ----
static __device__ __forceinline__ void emit_frag(_Float16* dst, const float* W,
                                                 int N, int KC, int idx) {
  int ti  = idx >> 9;
  int rem = idx & 511;
  int l = rem >> 4, e = rem & 15;
  int nt = ti / KC, kc = ti - nt * KC;
  int k = kc * 32 + ((e & 8) << 1) + ((l >> 4) << 3) + (e & 7);
  int n = nt * 16 + (l & 15);
  dst[idx] = (_Float16)W[k * N + n];
}

__global__ void nmodel_prep(const float* __restrict__ Wf, const float* __restrict__ W1,
                            const float* __restrict__ W2, const float* __restrict__ W3,
                            const float* __restrict__ Wo, _Float16* __restrict__ ws) {
  int i = blockIdx.x * blockDim.x + threadIdx.x;
  if      (i < 16384)  emit_frag(ws,          Wf, 128, 4, i);           // 4x8   tiles
  else if (i < 65536)  emit_frag(ws + 16384,  W1, 256, 6, i - 16384);   // 6x16  tiles
  else if (i < 131072) emit_frag(ws + 65536,  W2, 256, 8, i - 65536);   // 8x16  tiles
  else if (i < 163840) emit_frag(ws + 131072, W3, 128, 8, i - 131072);  // 8x8   tiles
  else if (i < 172032) emit_frag(ws + 163840, Wo, 64,  4, i - 163840);  // 4x4   tiles
}

// ---- main persistent kernel: one wave per 16 batch rows ----
__global__ __launch_bounds__(32)
__attribute__((amdgpu_num_vgpr(256)))
void nmodel_main(
    const float* __restrict__ t,  const float* __restrict__ x,
    const float* __restrict__ bf, const float* __restrict__ b1_in,
    const float* __restrict__ b2_in, const float* __restrict__ b3_in,
    const float* __restrict__ bo_in, const _Float16* __restrict__ wfrag,
    float* __restrict__ out)
{
  __shared__ __align__(32) _Float16 ldsWf[16384];   // 32 KB: Wf B-fragments
  __shared__ __align__(32) _Float16 ldsA[16 * 256]; // 8 KB: A-operand staging

  const int L  = threadIdx.x;
  const int ls = L & 15;
  const int HL = L >> 4;
  const int gbase = blockIdx.x * 16;

  // stage Wf fragments into LDS (1 KB per lane)
  for (int i = L * 16; i < 16384; i += 512)
    *(h16x16*)&ldsWf[i] = *(const h16x16*)&wfrag[i];
  __syncthreads();

  float bf_l[8];
#pragma unroll
  for (int nt = 0; nt < 8; nt++) bf_l[nt] = bf[nt * 16 + ls];

  v8f h[8];
#pragma unroll
  for (int nt = 0; nt < 8; nt++) h[nt] = splat8(0.f);

  // invariant: ldsA holds current h (A staging) at step top and substep ends
#pragma unroll
  for (int nt = 0; nt < 8; nt++) storeD(ldsA, nt, HL, ls, h[nt]);
  __syncthreads();

  const float dt  = 1.0f / NSTEPS;
  const float dt6 = dt / 6.0f;
  const float dt3 = dt / 3.0f;

#pragma unroll 1
  for (int step = 0; step < NT; ++step) {
    // per-iteration opaque offset: nothing weight/bias-related is loop-invariant,
    // but pointers still trace to kernargs -> loads stay global_load
    const int z0 = opaque0();
    const _Float16* gW1 = wfrag + 16384  + z0;
    const _Float16* gW2 = wfrag + 65536  + z0;
    const _Float16* gW3 = wfrag + 131072 + z0;
    const _Float16* gWo = wfrag + 163840 + z0;
    const float*    b1  = b1_in + z0;
    const float*    b2  = b2_in + z0;
    const float*    b3  = b3_in + z0;
    const float*    bo  = bo_in + z0;

    // ---- out[step] = h_prev @ Wo + bo   (ldsA already holds h_prev) ----
    {
      h16x16 ach[4];
#pragma unroll
      for (int c = 0; c < 4; c++) ach[c] = loadA(ldsA + ls * 256, c, HL);
      SB();
#pragma unroll
      for (int nt = 0; nt < 4; nt++) {
        v8f acc = splat8(bo[nt * 16 + ls]);
#pragma unroll
        for (int c = 0; c < 4; c++)
          acc = wmma16(ach[c], loadB(gWo, nt * 4 + c, L), acc);
        float* op = out + ((size_t)step * NB + gbase + 8 * HL) * MODES + nt * 16 + ls;
#pragma unroll
        for (int r = 0; r < 8; r++) op[(size_t)r * MODES] = acc[r];
        if (nt & 1) SB();
      }
    }

    float cr[8];
#pragma unroll
    for (int r = 0; r < 8; r++) cr[r] = t[(size_t)step * NB + gbase + 8 * HL + r];

    if (step + 1 < NT)
      __builtin_prefetch(x + ((size_t)(step + 1) * NB + gbase + ls) * MODES, 0, 1);

    // ---- RK4: 8 substeps x 4 fused stages ----
    {
      v8f hn[8];
#pragma unroll 1
      for (int s = 0; s < NSTEPS; ++s) {
        rkstage<true,  false>(ldsA, ldsWf, h, hn, dt6, 0.5f * dt, cr, bf_l, HL, ls, L); // k1
        rkstage<false, false>(ldsA, ldsWf, h, hn, dt3, 0.5f * dt, cr, bf_l, HL, ls, L); // k2
        rkstage<false, false>(ldsA, ldsWf, h, hn, dt3, dt,        cr, bf_l, HL, ls, L); // k3
        rkstage<false, true >(ldsA, ldsWf, h, hn, dt6, 0.f,       cr, bf_l, HL, ls, L); // k4
      }
    }

    // ---- MLP: z = [h_ode, x]; tanh(zW1+b1); tanh(.W2+b2); .W3+b3 ----
    // ldsA holds h_ode (cols 0..127)
    h16x16 zch[6];
#pragma unroll
    for (int c = 0; c < 4; c++) zch[c] = loadA(ldsA + ls * 256, c, HL);
    {
      const float* xr = x + ((size_t)step * NB + gbase + ls) * MODES;
#pragma unroll
      for (int c = 0; c < 2; c++) {
        v4f a0 = *(const v4f*)(xr + c * 32 + 8 * HL);
        v4f a1 = *(const v4f*)(xr + c * 32 + 8 * HL + 4);
        v4f c0 = *(const v4f*)(xr + c * 32 + 16 + 8 * HL);
        v4f c1 = *(const v4f*)(xr + c * 32 + 16 + 8 * HL + 4);
        h16x16 v;
#pragma unroll
        for (int i = 0; i < 4; i++) {
          v[i]      = (_Float16)a0[i];
          v[i + 4]  = (_Float16)a1[i];
          v[i + 8]  = (_Float16)c0[i];
          v[i + 12] = (_Float16)c1[i];
        }
        zch[4 + c] = v;
      }
    }
    __syncthreads();
    SB();
#pragma unroll
    for (int nt = 0; nt < 16; nt++) {                 // layer 1 (192 -> 256)
      v8f acc = splat8(b1[nt * 16 + ls]);
#pragma unroll
      for (int c = 0; c < 6; c++)
        acc = wmma16(zch[c], loadB(gW1, nt * 6 + c, L), acc);
#pragma unroll
      for (int r = 0; r < 8; r++) acc[r] = tanh_f(acc[r]);
      storeD(ldsA, nt, HL, ls, acc);
      if (nt & 1) SB();
    }
    __syncthreads();
    {
      h16x16 z2[8];
#pragma unroll
      for (int c = 0; c < 8; c++) z2[c] = loadA(ldsA + ls * 256, c, HL);
      __syncthreads();
      SB();
#pragma unroll
      for (int nt = 0; nt < 16; nt++) {               // layer 2 (256 -> 256)
        v8f acc = splat8(b2[nt * 16 + ls]);
#pragma unroll
        for (int c = 0; c < 8; c++)
          acc = wmma16(z2[c], loadB(gW2, nt * 8 + c, L), acc);
#pragma unroll
        for (int r = 0; r < 8; r++) acc[r] = tanh_f(acc[r]);
        storeD(ldsA, nt, HL, ls, acc);
        if (nt & 1) SB();
      }
    }
    __syncthreads();
    {
      h16x16 z3[8];
#pragma unroll
      for (int c = 0; c < 8; c++) z3[c] = loadA(ldsA + ls * 256, c, HL);
      __syncthreads();
      SB();
#pragma unroll
      for (int nt = 0; nt < 8; nt++) {                // layer 3 (256 -> 128) -> new h
        v8f acc = splat8(b3[nt * 16 + ls]);
#pragma unroll
        for (int c = 0; c < 8; c++)
          acc = wmma16(z3[c], loadB(gW3, nt * 8 + c, L), acc);
        h[nt] = acc;
        storeD(ldsA, nt, HL, ls, acc);                // restore ldsA = h invariant
        if (nt & 1) SB();
      }
    }
    __syncthreads();
  }
}

extern "C" void kernel_launch(void* const* d_in, const int* in_sizes, int n_in,
                              void* d_out, int out_size, void* d_ws, size_t ws_size,
                              hipStream_t stream) {
  const float* t  = (const float*)d_in[0];
  const float* x  = (const float*)d_in[1];
  const float* Wf = (const float*)d_in[2];
  const float* bf = (const float*)d_in[3];
  const float* W1 = (const float*)d_in[4];
  const float* b1 = (const float*)d_in[5];
  const float* W2 = (const float*)d_in[6];
  const float* b2 = (const float*)d_in[7];
  const float* W3 = (const float*)d_in[8];
  const float* b3 = (const float*)d_in[9];
  const float* Wo = (const float*)d_in[10];
  const float* bo = (const float*)d_in[11];
  _Float16* wfrag = (_Float16*)d_ws;   // needs 344 KB of workspace

  nmodel_prep<<<672, 256, 0, stream>>>(Wf, W1, W2, W3, Wo, wfrag);
  nmodel_main<<<64, 32, 0, stream>>>(t, x, bf, b1, b2, b3, bo, wfrag, (float*)d_out);
}
